// LeakyRNN_50792283243103
// MI455X (gfx1250) — compile-verified, matrix operand order checked by reading, not run on hardware
//
#include <hip/hip_runtime.h>

// ---------------------------------------------------------------------------
// Leaky-integration RNN for MI455X (gfx1250, wave32, WMMA).
//   Pass 0: pack Win/Wr into bf16 WMMA B-fragment layout in d_ws (~768 KB).
//   Pass 1: U[t] = x[t] @ Win + bias  -> stored straight into d_out (parallel GEMM).
//   Pass 2: sequential scan, one WG per 16-row batch tile (no inter-WGP sync
//           needed: the recurrence is independent per batch row). h kept in
//           fp32 registers (C fragments), mirrored as bf16 A-fragments in
//           double-buffered LDS for the next step's h @ Wr.
// ---------------------------------------------------------------------------

#define SEQ   1024
#define BATCH 128
#define DIN   256
#define HID   512
#define ROWS  (SEQ * BATCH)

typedef __attribute__((ext_vector_type(16))) __bf16 v16bf;
typedef __attribute__((ext_vector_type(8)))  float  v8f;

__device__ __forceinline__ __bf16 f2bf(float f) {
  // round-to-nearest-even fp32 -> bf16
  unsigned u = __builtin_bit_cast(unsigned, f);
  unsigned r = u + 0x7FFFu + ((u >> 16) & 1u);
  unsigned short h = (unsigned short)(r >> 16);
  return __builtin_bit_cast(__bf16, h);
}

// Packed B-fragment layout: elem index = ((nt*KT + kt)*512 + l*16 + j)
//   lane l of tile (nt,kt):  n = nt*16 + (l&15),  k = kt*32 + (l>>4)*16 + j
// so each lane's 16 bf16 (32 bytes) are contiguous -> single v16bf load.
__global__ void __launch_bounds__(256)
pack_weights(const float* __restrict__ win, const float* __restrict__ wr,
             __bf16* __restrict__ winp, __bf16* __restrict__ wrp) {
  int idx = blockIdx.x * blockDim.x + threadIdx.x;
  if (idx < 32 * 8 * 512) {                 // Win: NT=32 (512/16), KT=8 (256/32)
    int j = idx & 15, l = (idx >> 4) & 31, tile = idx >> 9;
    int kt = tile & 7, nt = tile >> 3;
    int k = kt * 32 + ((l >> 4) << 4) + j;
    int n = nt * 16 + (l & 15);
    winp[idx] = f2bf(win[k * HID + n]);
  }
  idx -= 32 * 8 * 512;
  if (idx >= 0 && idx < 32 * 16 * 512) {    // Wr: NT=32, KT=16 (512/32)
    int j = idx & 15, l = (idx >> 4) & 31, tile = idx >> 9;
    int kt = tile & 15, nt = tile >> 4;
    int k = kt * 32 + ((l >> 4) << 4) + j;
    int n = nt * 16 + (l & 15);
    wrp[idx] = f2bf(wr[k * HID + n]);
  }
}

// Pass 1: U = x @ Win + bias, written to out[t]. One block = one 16-row tile,
// 8 waves each own 64 output columns. x staged once per block as bf16 in LDS.
__global__ void __launch_bounds__(256)
input_gemm(const float* __restrict__ x, const __bf16* __restrict__ winp,
           const float* __restrict__ bias, float* __restrict__ out) {
  __shared__ __bf16 xs[16 * 272];           // 272 = 256 + pad (544B rows, bank-staggered)
  const int tid = threadIdx.x;
  const int rt  = blockIdx.x;               // row tile 0..8191
  const int w   = tid >> 5, l = tid & 31;
  const int lr  = l & 15,  lh = l >> 4;

  const float* xrow = x + (size_t)rt * 16 * DIN;
  for (int i = tid; i < 16 * DIN; i += 256) {
    int r = i >> 8, k = i & 255;
    xs[r * 272 + k] = f2bf(xrow[i]);
  }
  __syncthreads();

  v8f acc[4] = {};
  for (int kt = 0; kt < 8; ++kt) {
    const v16bf a = *reinterpret_cast<const v16bf*>(&xs[lr * 272 + kt * 32 + lh * 16]);
#pragma unroll
    for (int ct = 0; ct < 4; ++ct) {
      const int nt = w * 4 + ct;
      const v16bf b = *reinterpret_cast<const v16bf*>(winp + ((size_t)(nt * 8 + kt) * 512 + l * 16));
      acc[ct] = __builtin_amdgcn_wmma_f32_16x16x32_bf16(false, a, false, b,
                                                        (short)0, acc[ct], false, false);
    }
  }
#pragma unroll
  for (int ct = 0; ct < 4; ++ct) {
    const int col = w * 64 + ct * 16 + lr;
    const float bv = bias[col];
#pragma unroll
    for (int r = 0; r < 8; ++r) {
      const int row = rt * 16 + r + lh * 8;
      out[(size_t)row * HID + col] = acc[ct][r] + bv;
    }
  }
}

// Pass 2: sequential scan. Block g owns batch rows [g*16, g*16+16); wave w owns
// hidden columns [w*64, w*64+64). h state lives in fp32 C-fragment registers;
// bf16 copy double-buffered in LDS as row-major A-source. One barrier per step.
__global__ void __launch_bounds__(256)
recurrence(const __bf16* __restrict__ wrp, float* __restrict__ out) {
  __shared__ __bf16 hb[2][16 * 528];        // 528 = 512 + pad (1056B rows, 32B-aligned frags)
  const int tid = threadIdx.x;
  const int g   = blockIdx.x;
  const int w   = tid >> 5, l = tid & 31;
  const int lr  = l & 15,  lh = l >> 4;
  const int colbase = w * 64;

  for (int i = tid; i < 2 * 16 * 528; i += 256)
    (&hb[0][0])[i] = __builtin_bit_cast(__bf16, (unsigned short)0);
  __syncthreads();

  float hc[4][8];
#pragma unroll
  for (int ct = 0; ct < 4; ++ct)
#pragma unroll
    for (int r = 0; r < 8; ++r) hc[ct][r] = 0.f;

  for (int t = 0; t < SEQ; ++t) {
    const int rb = t & 1, wb = rb ^ 1;
    float* outt = out + (size_t)t * (BATCH * HID) + (size_t)g * 16 * HID;

    // acc starts as U_t (= x@Win + bias), read before being overwritten below.
    v8f acc[4];
#pragma unroll
    for (int ct = 0; ct < 4; ++ct) {
      const int col = colbase + ct * 16 + lr;
#pragma unroll
      for (int r = 0; r < 8; ++r)
        acc[ct][r] = outt[(size_t)(r + lh * 8) * HID + col];
    }
    if (t + 1 < SEQ)  // pull next step's U tile toward L2/L0 (global_prefetch_b8)
      __builtin_prefetch(outt + (size_t)(BATCH * HID) + (size_t)(lh * 8) * HID + colbase + lr, 0, 1);

    // acc += h_{t-1} @ Wr
    for (int kt = 0; kt < 16; ++kt) {
      const v16bf a = *reinterpret_cast<const v16bf*>(&hb[rb][lr * 528 + kt * 32 + lh * 16]);
#pragma unroll
      for (int ct = 0; ct < 4; ++ct) {
        const int nt = (colbase >> 4) + ct;
        const v16bf b = *reinterpret_cast<const v16bf*>(wrp + ((size_t)(nt * 16 + kt) * 512 + l * 16));
        acc[ct] = __builtin_amdgcn_wmma_f32_16x16x32_bf16(false, a, false, b,
                                                          (short)0, acc[ct], false, false);
      }
    }

    // leaky update: h = h + (tanh(z) - h)/tau ; write fp32 out + bf16 LDS mirror
#pragma unroll
    for (int ct = 0; ct < 4; ++ct) {
      const int col = colbase + ct * 16 + lr;
#pragma unroll
      for (int r = 0; r < 8; ++r) {
        const int m = r + lh * 8;
        const float hn = hc[ct][r] + 0.1f * (tanhf(acc[ct][r]) - hc[ct][r]);
        hc[ct][r] = hn;
        outt[(size_t)m * HID + col] = hn;
        hb[wb][m * 528 + col] = f2bf(hn);
      }
    }
    __syncthreads();   // double buffer -> single barrier per step suffices
  }
}

extern "C" void kernel_launch(void* const* d_in, const int* in_sizes, int n_in,
                              void* d_out, int out_size, void* d_ws, size_t ws_size,
                              hipStream_t stream) {
  const float* x    = (const float*)d_in[0];
  const float* win  = (const float*)d_in[1];
  const float* wr   = (const float*)d_in[2];
  const float* bias = (const float*)d_in[3];
  float* out = (float*)d_out;

  __bf16* winp = (__bf16*)d_ws;             // 131072 elems = 256 KB
  __bf16* wrp  = winp + 32 * 8 * 512;       // 262144 elems = 512 KB (ws use: 768 KB)

  pack_weights<<<1536, 256, 0, stream>>>(win, wr, winp, wrp);
  input_gemm <<<ROWS / 16, 256, 0, stream>>>(x, winp, bias, out);
  recurrence <<<BATCH / 16, 256, 0, stream>>>(wrp, out);
}
